// HouseholdDiffusionGNN_11682311045871
// MI455X (gfx1250) — matrix operand order repaired and checked by reading, not compiled, not run on hardware
//
#include <hip/hip_runtime.h>
#include <hip/hip_bf16.h>

#define NNODES 50000
#define NEDGES 800000
#define NBATCH 5000
#define ETOT   (NEDGES + NNODES)
#define HDIM   128
#define INDIM  288
#define EDIM   16
#define NDIM   32
#define HHD    64
#define NLAYER 3

typedef __attribute__((ext_vector_type(16))) __bf16 v16bf;
typedef __attribute__((ext_vector_type(8)))  float  v8f;

union F16Frag { v16bf v; uint4 q[2]; };

__device__ __forceinline__ unsigned short f2bf(float f) {
    unsigned int u = __float_as_uint(f);
    u += 0x7FFFu + ((u >> 16) & 1u);   // round-to-nearest-even
    return (unsigned short)(u >> 16);
}

__device__ __forceinline__ float siluf(float v) {
    return v / (1.0f + __expf(-v));
}

__device__ __forceinline__ void atomicMaxF(float* addr, float val) {
    // monotonic float-bits trick: signed max for >=0, unsigned min for <0
    if (val >= 0.0f) atomicMax((int*)addr, __float_as_int(val));
    else             atomicMin((unsigned int*)addr, __float_as_uint(val));
}

// ---------------------------------------------------------------- utilities
__global__ void fill_f32(float* __restrict__ p, float v, int n) {
    int i = blockIdx.x * blockDim.x + threadIdx.x;
    if (i < n) p[i] = v;
}

__global__ void cvt_bf16(const float* __restrict__ in, unsigned short* __restrict__ o, int n) {
    int i = blockIdx.x * blockDim.x + threadIdx.x;
    if (i < n) o[i] = f2bf(in[i]);
}

// ------------------------------------------------- bf16 WMMA GEMM, C=f32
// A: [M,K] row-major bf16, B: [K,Nc] row-major bf16, C: [M,Nc] f32 (ldc),
// bias added per-column. One wave computes one 16x32 tile (two 16x16 WMMA
// accumulators sharing one A fragment). K % 32 == 0, Nc % 32 == 0.
// OOB M-rows: A row pointer is clamped to row 0 (loads always legal); the
// resulting garbage output rows are simply not stored.
__global__ __launch_bounds__(256) void gemm_bf16(
    const unsigned short* __restrict__ A, int lda,
    const unsigned short* __restrict__ B, int ldb,
    const float* __restrict__ bias,
    float* __restrict__ C, int ldc,
    int M, int K, int Nc)
{
    int wave = (blockIdx.x * 256 + threadIdx.x) >> 5;
    int lane = threadIdx.x & 31;
    int nT = Nc >> 5;                 // 32-column super-tiles
    int mT = (M + 15) >> 4;
    if (wave >= mT * nT) return;
    int mt = wave / nT, nt = wave % nT;

    int row  = mt * 16 + (lane & 15);
    int rowc = row < M ? row : 0;                 // clamp: OOB rows -> garbage, never stored
    int kb   = (lane >> 4) << 3;                  // A: lanes 16-31 start at K+8
    const unsigned short* Ap  = A + (size_t)rowc * lda + kb;
    const unsigned short* Bp0 = B + (size_t)kb * ldb + nt * 32 + (lane & 15);   // B: column = lane%16, K from kb
    const unsigned short* Bp1 = Bp0 + 16;

    v8f acc0 = {0.f, 0.f, 0.f, 0.f, 0.f, 0.f, 0.f, 0.f};
    v8f acc1 = {0.f, 0.f, 0.f, 0.f, 0.f, 0.f, 0.f, 0.f};
    for (int k0 = 0; k0 < K; k0 += 32) {
        F16Frag a, b0, b1;
        a.q[0] = *(const uint4*)(Ap + k0);        // K = k0+kb .. +7
        a.q[1] = *(const uint4*)(Ap + k0 + 16);   // K = k0+kb+16 .. +23
        size_t boff = (size_t)k0 * ldb;
#pragma unroll
        for (int s = 0; s < 8; ++s) {
            b0.v[s]     = __builtin_bit_cast(__bf16, Bp0[boff + (size_t)s * ldb]);
            b0.v[s + 8] = __builtin_bit_cast(__bf16, Bp0[boff + (size_t)(s + 16) * ldb]);
            b1.v[s]     = __builtin_bit_cast(__bf16, Bp1[boff + (size_t)s * ldb]);
            b1.v[s + 8] = __builtin_bit_cast(__bf16, Bp1[boff + (size_t)(s + 16) * ldb]);
        }
        acc0 = __builtin_amdgcn_wmma_f32_16x16x32_bf16(
            false, a.v, false, b0.v, (short)0, acc0, false, false);
        acc1 = __builtin_amdgcn_wmma_f32_16x16x32_bf16(
            false, a.v, false, b1.v, (short)0, acc1, false, false);
    }
    int n0 = nt * 32 + (lane & 15);
    int n1 = n0 + 16;
    float bv0 = bias ? bias[n0] : 0.0f;
    float bv1 = bias ? bias[n1] : 0.0f;
    int mbase = mt * 16 + ((lane >> 4) << 3);
#pragma unroll
    for (int v = 0; v < 8; ++v) {
        int r = mbase + v;
        if (r < M) {
            C[(size_t)r * ldc + n0] = acc0[v] + bv0;
            C[(size_t)r * ldc + n1] = acc1[v] + bv1;
        }
    }
}

// --------------------------------------------------------- small MLP pieces
__global__ void time_hidden(const int* __restrict__ t,
                            const float* __restrict__ tW1, const float* __restrict__ tb1,
                            unsigned short* __restrict__ Th, int total) {
    int i = blockIdx.x * blockDim.x + threadIdx.x;
    if (i >= total) return;
    int j = i & (HDIM - 1);
    int n = i >> 7;
    float h = (float)t[n] * tW1[j] + tb1[j];
    Th[i] = f2bf(siluf(h));
}

// per-row LayerNorm(128) + SiLU, f32 -> f32
__global__ __launch_bounds__(128) void ln_silu(
    const float* __restrict__ X, const float* __restrict__ g,
    const float* __restrict__ b, float* __restrict__ Y, int R)
{
    int r = blockIdx.x;
    if (r >= R) return;
    int j = threadIdx.x;
    float x = X[(size_t)r * HDIM + j];
    float s1 = x, s2 = x * x;
#pragma unroll
    for (int off = 16; off > 0; off >>= 1) {
        s1 += __shfl_xor(s1, off, 32);
        s2 += __shfl_xor(s2, off, 32);
    }
    __shared__ float sm1[4], sm2[4];
    int wid = j >> 5, lane = j & 31;
    if (lane == 0) { sm1[wid] = s1; sm2[wid] = s2; }
    __syncthreads();
    float t1 = sm1[0] + sm1[1] + sm1[2] + sm1[3];
    float t2 = sm2[0] + sm2[1] + sm2[2] + sm2[3];
    float mu  = t1 * (1.0f / HDIM);
    float var = t2 * (1.0f / HDIM) - mu * mu;
    float y = (x - mu) * rsqrtf(var + 1e-5f) * g[j] + b[j];
    Y[(size_t)r * HDIM + j] = siluf(y);
}

// concat: X[:,0:32]=x_t ; X[:,160:288]=h[batch] (cols 32:160 written by GEMM)
__global__ void build_x(const float* __restrict__ x_t, const float* __restrict__ h,
                        const int* __restrict__ batch, float* __restrict__ X, int total) {
    int i = blockIdx.x * blockDim.x + threadIdx.x;
    if (i >= total) return;
    int c = i % 160;
    int n = i / 160;
    if (c < NDIM) X[(size_t)n * INDIM + c] = x_t[(size_t)n * NDIM + c];
    else {
        int j = c - NDIM;
        X[(size_t)n * INDIM + (NDIM + HDIM) + j] = h[(size_t)batch[n] * HDIM + j];
    }
}

// ----------------------------------------------------------- edge-attr mean
__global__ void ea_accum(const float* __restrict__ ea, float* __restrict__ acc, int E) {
    __shared__ float s[EDIM];
    if (threadIdx.x < EDIM) s[threadIdx.x] = 0.0f;
    __syncthreads();
    float loc[EDIM];
#pragma unroll
    for (int k = 0; k < EDIM; ++k) loc[k] = 0.0f;
    for (int e = blockIdx.x * blockDim.x + threadIdx.x; e < E; e += gridDim.x * blockDim.x) {
#pragma unroll
        for (int k = 0; k < EDIM; ++k) loc[k] += ea[(size_t)e * EDIM + k];
    }
#pragma unroll
    for (int k = 0; k < EDIM; ++k) atomicAdd(&s[k], loc[k]);
    __syncthreads();
    if (threadIdx.x < EDIM) atomicAdd(&acc[threadIdx.x], s[threadIdx.x]);
}

__global__ void ea_finalize(float* __restrict__ acc, float inv) {
    if (threadIdx.x < EDIM) acc[threadIdx.x] *= inv;
}

// -------------------------------------------------------------- edge passes
// pass 1: score[e] = att . leaky_relu(xl[src]+xr[dst]+ea@We), atomic max per dst
__global__ __launch_bounds__(256) void edge_score(
    const int* __restrict__ ei, const float* __restrict__ eattr,
    const float* __restrict__ eamean,
    const float* __restrict__ XL, const float* __restrict__ XR,
    const float* __restrict__ We, const float* __restrict__ att,
    float* __restrict__ SC, float* __restrict__ SMAX)
{
    __shared__ float sWe[EDIM * HDIM];
    __shared__ float sAtt[HDIM];
    for (int i = threadIdx.x; i < EDIM * HDIM; i += 256) sWe[i] = We[i];
    if (threadIdx.x < HDIM) sAtt[threadIdx.x] = att[threadIdx.x];
    __syncthreads();

    int e = blockIdx.x * 8 + (threadIdx.x >> 5);
    if (e >= ETOT) return;
    int lane = threadIdx.x & 31;
    int s, d; const float* ea;
    if (e < NEDGES) { s = ei[e]; d = ei[NEDGES + e]; ea = eattr + (size_t)e * EDIM; }
    else            { s = e - NEDGES; d = s;          ea = eamean; }

    int jb = lane * 4;
    float4 xl = *(const float4*)(XL + (size_t)s * HDIM + jb);
    float4 xr = *(const float4*)(XR + (size_t)d * HDIM + jb);

    float w0 = 0.f, w1 = 0.f, w2 = 0.f, w3 = 0.f;
#pragma unroll
    for (int k = 0; k < EDIM; ++k) {
        float a = ea[k];
        const float* wr = sWe + k * HDIM + jb;
        w0 += a * wr[0]; w1 += a * wr[1]; w2 += a * wr[2]; w3 += a * wr[3];
    }
    float p = 0.f, v;
    v = xl.x + xr.x + w0; v = v > 0.f ? v : 0.2f * v; p += sAtt[jb + 0] * v;
    v = xl.y + xr.y + w1; v = v > 0.f ? v : 0.2f * v; p += sAtt[jb + 1] * v;
    v = xl.z + xr.z + w2; v = v > 0.f ? v : 0.2f * v; p += sAtt[jb + 2] * v;
    v = xl.w + xr.w + w3; v = v > 0.f ? v : 0.2f * v; p += sAtt[jb + 3] * v;
#pragma unroll
    for (int off = 16; off > 0; off >>= 1) p += __shfl_xor(p, off, 32);
    if (lane == 0) { SC[e] = p; atomicMaxF(&SMAX[d], p); }
}

// pass 2: ex = exp(score - smax[dst]); denom[dst] += ex
__global__ void edge_exp(const int* __restrict__ ei,
                         float* __restrict__ SC, const float* __restrict__ SMAX,
                         float* __restrict__ DEN)
{
    int e = blockIdx.x * blockDim.x + threadIdx.x;
    if (e >= ETOT) return;
    int d = (e < NEDGES) ? ei[NEDGES + e] : (e - NEDGES);
    float ex = __expf(SC[e] - SMAX[d]);
    SC[e] = ex;
    atomicAdd(&DEN[d], ex);
}

// pass 3: ACC[dst] += (ex/denom[dst]) * xl[src]
__global__ __launch_bounds__(256) void edge_aggr(
    const int* __restrict__ ei, const float* __restrict__ SC,
    const float* __restrict__ DEN, const float* __restrict__ XL,
    float* __restrict__ ACC)
{
    int e = blockIdx.x * 8 + (threadIdx.x >> 5);
    if (e >= ETOT) return;
    int lane = threadIdx.x & 31;
    int s, d;
    if (e < NEDGES) { s = ei[e]; d = ei[NEDGES + e]; }
    else            { s = e - NEDGES; d = s; }
    float alpha = SC[e] / (DEN[d] + 1e-16f);
    int jb = lane * 4;
    float4 xl = *(const float4*)(XL + (size_t)s * HDIM + jb);
    float* o = ACC + (size_t)d * HDIM + jb;
    atomicAdd(o + 0, alpha * xl.x);
    atomicAdd(o + 1, alpha * xl.y);
    atomicAdd(o + 2, alpha * xl.z);
    atomicAdd(o + 3, alpha * xl.w);
}

// x = silu(ACC + gb) -> bf16
__global__ void silu_eps(const float* __restrict__ ACC, const float* __restrict__ gb,
                         unsigned short* __restrict__ Xb, int total) {
    int i = blockIdx.x * blockDim.x + threadIdx.x;
    if (i >= total) return;
    float v = ACC[i] + gb[i & (HDIM - 1)];
    Xb[i] = f2bf(siluf(v));
}

// =============================================================== host side
extern "C" void kernel_launch(void* const* d_in, const int* in_sizes, int n_in,
                              void* d_out, int out_size, void* d_ws, size_t ws_size,
                              hipStream_t stream) {
    (void)in_sizes; (void)n_in; (void)out_size; (void)ws_size;
    const float* x_t   = (const float*)d_in[0];
    const int*   t     = (const int*)  d_in[1];
    const float* hh    = (const float*)d_in[2];
    const int*   ei    = (const int*)  d_in[3];
    const float* eattr = (const float*)d_in[4];
    const int*   batch = (const int*)  d_in[5];
    const float* tW1 = (const float*)d_in[6];  const float* tb1 = (const float*)d_in[7];
    const float* tW2 = (const float*)d_in[8];  const float* tb2 = (const float*)d_in[9];
    const float* hW1 = (const float*)d_in[10]; const float* hb1 = (const float*)d_in[11];
    const float* hg1 = (const float*)d_in[12]; const float* hB1 = (const float*)d_in[13];
    const float* hW2 = (const float*)d_in[14]; const float* hb2 = (const float*)d_in[15];
    const float* hg2 = (const float*)d_in[16]; const float* hB2 = (const float*)d_in[17];
    const float* Wl  = (const float*)d_in[18]; const float* bl  = (const float*)d_in[19];
    const float* Wr  = (const float*)d_in[20]; const float* br  = (const float*)d_in[21];
    const float* We  = (const float*)d_in[22]; const float* att = (const float*)d_in[23];
    const float* gb  = (const float*)d_in[24];
    const float* fcpW = (const float*)d_in[25]; const float* fcpb = (const float*)d_in[26];
    const float* npW1 = (const float*)d_in[27]; const float* npb1 = (const float*)d_in[28];
    const float* npg  = (const float*)d_in[29]; const float* npB  = (const float*)d_in[30];
    const float* npW2 = (const float*)d_in[31]; const float* npb2 = (const float*)d_in[32];
    float* out = (float*)d_out;

    // ---- workspace carve
    char* p = (char*)d_ws;
    auto alloc = [&](size_t bytes) -> char* {
        char* r = p; p += (bytes + 255) & ~(size_t)255; return r;
    };
    float*          Xf   = (float*)         alloc((size_t)NNODES * INDIM * 4);
    unsigned short* Xb   = (unsigned short*)alloc((size_t)NNODES * INDIM * 2);
    float*          XLb  = (float*)         alloc((size_t)NNODES * HDIM * 4);
    float*          XRb  = (float*)         alloc((size_t)NNODES * HDIM * 4);
    float*          ACC  = (float*)         alloc((size_t)NNODES * HDIM * 4);
    unsigned short* XCb  = (unsigned short*)alloc((size_t)NNODES * HDIM * 2);
    float*          SC   = (float*)         alloc((size_t)ETOT * 4);
    float*          SMAX = (float*)         alloc((size_t)NNODES * 4);
    float*          DEN  = (float*)         alloc((size_t)NNODES * 4);
    float*          HA   = (float*)         alloc((size_t)NBATCH * HDIM * 4);
    float*          HB   = (float*)         alloc((size_t)NBATCH * HDIM * 4);
    unsigned short* Hbf  = (unsigned short*)alloc((size_t)NBATCH * HDIM * 2);
    unsigned short* HHb  = (unsigned short*)alloc((size_t)NBATCH * HHD * 2);
    float*          EAM  = (float*)         alloc(EDIM * 4);
    unsigned short* Wl_b  = (unsigned short*)alloc((size_t)NLAYER * INDIM * HDIM * 2);
    unsigned short* Wr_b  = (unsigned short*)alloc((size_t)NLAYER * INDIM * HDIM * 2);
    unsigned short* tW2_b = (unsigned short*)alloc((size_t)HDIM * HDIM * 2);
    unsigned short* hW1_b = (unsigned short*)alloc((size_t)HHD * HDIM * 2);
    unsigned short* hW2_b = (unsigned short*)alloc((size_t)HDIM * HDIM * 2);
    unsigned short* fcpW_b= (unsigned short*)alloc((size_t)HDIM * INDIM * 2);
    unsigned short* npW1_b= (unsigned short*)alloc((size_t)HDIM * HDIM * 2);
    unsigned short* npW2_b= (unsigned short*)alloc((size_t)HDIM * NDIM * 2);

    auto cvt = [&](const float* in, unsigned short* o, int n) {
        cvt_bf16<<<(n + 255) / 256, 256, 0, stream>>>(in, o, n);
    };
    auto gemm = [&](const unsigned short* A, int lda, const unsigned short* Bm, int ldb,
                    const float* bias, float* C, int ldc, int M, int K, int Nc) {
        int tiles  = ((M + 15) / 16) * (Nc / 32);
        int blocks = (tiles + 7) / 8;
        gemm_bf16<<<blocks, 256, 0, stream>>>(A, lda, Bm, ldb, bias, C, ldc, M, K, Nc);
    };

    // ---- weight conversion to bf16
    cvt(Wl,   Wl_b,   NLAYER * INDIM * HDIM);
    cvt(Wr,   Wr_b,   NLAYER * INDIM * HDIM);
    cvt(tW2,  tW2_b,  HDIM * HDIM);
    cvt(hW1,  hW1_b,  HHD * HDIM);
    cvt(hW2,  hW2_b,  HDIM * HDIM);
    cvt(fcpW, fcpW_b, HDIM * INDIM);
    cvt(npW1, npW1_b, HDIM * HDIM);
    cvt(npW2, npW2_b, HDIM * NDIM);
    cvt(hh,   HHb,    NBATCH * HHD);

    // ---- edge_attr per-feature mean (self-loop fill)
    fill_f32<<<1, 64, 0, stream>>>(EAM, 0.0f, EDIM);
    ea_accum<<<256, 256, 0, stream>>>(eattr, EAM, NEDGES);
    ea_finalize<<<1, 64, 0, stream>>>(EAM, 1.0f / (float)NEDGES);

    // ---- time embedding: Th = silu(t*tW1+tb1) ; t_emb = Th @ tW2 + tb2 -> X[:,32:160]
    {
        int tot = NNODES * HDIM;
        time_hidden<<<(tot + 255) / 256, 256, 0, stream>>>(t, tW1, tb1, XCb, tot);
        gemm(XCb, HDIM, tW2_b, HDIM, tb2, Xf + NDIM, INDIM, NNODES, HDIM, HDIM);
    }

    // ---- household encoder: 2 x (Linear -> LN -> SiLU)
    gemm(HHb, HHD, hW1_b, HDIM, hb1, HA, HDIM, NBATCH, HHD, HDIM);
    ln_silu<<<NBATCH, 128, 0, stream>>>(HA, hg1, hB1, HB, NBATCH);
    cvt(HB, Hbf, NBATCH * HDIM);
    gemm(Hbf, HDIM, hW2_b, HDIM, hb2, HA, HDIM, NBATCH, HDIM, HDIM);
    ln_silu<<<NBATCH, 128, 0, stream>>>(HA, hg2, hB2, HB, NBATCH);

    // ---- X = concat(x_t, t_emb, h[batch]) ; bf16 copy
    {
        int tot = NNODES * 160;
        build_x<<<(tot + 255) / 256, 256, 0, stream>>>(x_t, HB, batch, Xf, tot);
        cvt(Xf, Xb, NNODES * INDIM);
    }

    const float NEGINF = -__builtin_inff();
    // ---- GATv2 layers
    for (int l = 0; l < NLAYER; ++l) {
        gemm(Xb, INDIM, Wl_b + (size_t)l * INDIM * HDIM, HDIM, bl + l * HDIM,
             XLb, HDIM, NNODES, INDIM, HDIM);
        gemm(Xb, INDIM, Wr_b + (size_t)l * INDIM * HDIM, HDIM, br + l * HDIM,
             XRb, HDIM, NNODES, INDIM, HDIM);

        fill_f32<<<(NNODES * HDIM + 255) / 256, 256, 0, stream>>>(ACC, 0.0f, NNODES * HDIM);
        fill_f32<<<(NNODES + 255) / 256, 256, 0, stream>>>(SMAX, NEGINF, NNODES);
        fill_f32<<<(NNODES + 255) / 256, 256, 0, stream>>>(DEN, 0.0f, NNODES);

        int eb8 = (ETOT + 7) / 8;
        edge_score<<<eb8, 256, 0, stream>>>(ei, eattr, EAM, XLb, XRb,
                                            We + (size_t)l * EDIM * HDIM,
                                            att + l * HDIM, SC, SMAX);
        edge_exp<<<(ETOT + 255) / 256, 256, 0, stream>>>(ei, SC, SMAX, DEN);
        edge_aggr<<<eb8, 256, 0, stream>>>(ei, SC, DEN, XLb, ACC);

        silu_eps<<<(NNODES * HDIM + 255) / 256, 256, 0, stream>>>(
            ACC, gb + l * HDIM, XCb, NNODES * HDIM);

        if (l < NLAYER - 1) {
            gemm(XCb, HDIM, fcpW_b, INDIM, fcpb, Xf, INDIM, NNODES, HDIM, INDIM);
            cvt(Xf, Xb, NNODES * INDIM);
        }
    }

    // ---- node projection: Linear -> LN -> SiLU -> Linear
    gemm(XCb, HDIM, npW1_b, HDIM, npb1, XLb, HDIM, NNODES, HDIM, HDIM);
    ln_silu<<<NNODES, 128, 0, stream>>>(XLb, npg, npB, XRb, NNODES);
    cvt(XRb, XCb, NNODES * HDIM);
    gemm(XCb, HDIM, npW2_b, NDIM, npb2, out, NDIM, NNODES, HDIM, NDIM);
}
